// Wasserstain_62466004353026
// MI455X (gfx1250) — matrix operand order repaired and checked by reading, not compiled, not run on hardware
//
#include <hip/hip_runtime.h>
#include <hip/hip_bf16.h>

// ---------------------------------------------------------------------------
// Fused SCAN/Sinkhorn similarity for MI455X (gfx1250, wave32, WMMA).
//   Phase A: convert f32 inputs -> f16, zero-padded to WMMA-friendly tiles.
//   Phase B: one block per (i,c) pair: f16 WMMA GEMM (48x64x1024) -> LDS,
//            then masked Sinkhorn (3 iters) + weighted reduction -> out[i,c].
// ---------------------------------------------------------------------------

typedef _Float16 f16x4 __attribute__((ext_vector_type(4)));
typedef _Float16 f16x8 __attribute__((ext_vector_type(8)));
typedef _Float16 v16h  __attribute__((ext_vector_type(16)));
typedef float    v8f   __attribute__((ext_vector_type(8)));

#define BI 128
#define BT 128
#define RR 36
#define WW 50
#define DD 1024
#define RP 48   // padded rows (3 x 16 M-tiles)
#define WP 64   // padded cols (4 x 16 N-tiles)
#define LP 65   // LDS row pitch (bank-conflict free)
#define LAMB_INV 20.0f
#define EPSV 1e-6f

#define SHUF16(a, b) __builtin_shufflevector((a), (b), 0,1,2,3,4,5,6,7,8,9,10,11,12,13,14,15)

// -------- Phase A: f32 -> f16 with row padding (zeros) ----------------------
__global__ void __launch_bounds__(256)
cvt_pad_f16(const float* __restrict__ src, _Float16* __restrict__ dst,
            int src_rows, int pad_rows) {
    long idx4 = (long)blockIdx.x * blockDim.x + threadIdx.x;
    long total4 = (long)BI * pad_rows * (DD / 4);
    if (idx4 >= total4) return;
    long e = idx4 * 4;                // element index in padded layout
    int  k = (int)(e & (DD - 1));
    long row = e >> 10;               // i*pad_rows + r
    int  r = (int)(row % pad_rows);
    long i = row / pad_rows;
    f16x4 o;
    if (r < src_rows) {
        const float4 v = *(const float4*)(src + (((i * src_rows + r) << 10) + k));
        o[0] = (_Float16)v.x; o[1] = (_Float16)v.y;
        o[2] = (_Float16)v.z; o[3] = (_Float16)v.w;
    } else {
        o[0] = (_Float16)0.f; o[1] = (_Float16)0.f;
        o[2] = (_Float16)0.f; o[3] = (_Float16)0.f;
    }
    *(f16x4*)(dst + e) = o;
}

// -------- Phase B: fused WMMA GEMM + Sinkhorn per (i,c) pair ----------------
__global__ void __launch_bounds__(128)
sinkhorn_wmma(const _Float16* __restrict__ A16,   // [BI][RP][DD] f16 (padded)
              const _Float16* __restrict__ B16,   // [BT][WP][DD] f16 (padded)
              const int* __restrict__ img_lens,
              const int* __restrict__ cap_lens,
              float* __restrict__ out) {
    __shared__ float S[RP * LP];
    __shared__ float P[RP * LP];
    __shared__ float red[128];
    __shared__ float rs[RP];
    __shared__ float cs[WP];

    const int c    = blockIdx.x;
    const int i    = blockIdx.y;
    const int tid  = threadIdx.x;
    const int lane = tid & 31;
    const int wv   = tid >> 5;        // wave id 0..3 -> N tile
    const int m    = lane & 15;
    const int hf   = lane >> 4;       // lane half (K grouping)

    // ---- GEMM: D[48x64] = A(48x1024) * B^T(1024x64), f16 in / f32 acc ----
    // A 16x32 layout: lane<16 holds K{0..7,16..23}, lane>=16 holds K{8..15,24..31}
    // B 32x16 layout: lane holds column N=lane%16, lane<16 holds K{0..7,16..23}, lane>=16 holds K{8..15,24..31}
    const _Float16* Ab  = A16 + (size_t)i * (RP * DD);
    const _Float16* Ar0 = Ab + (size_t)m * DD + 8 * hf;
    const _Float16* Ar1 = Ar0 + 16 * DD;
    const _Float16* Ar2 = Ar0 + 32 * DD;
    const _Float16* Bp  = B16 + (size_t)c * (WP * DD)
                              + (size_t)(16 * wv + m) * DD + 8 * hf;

    v8f acc0 = {}, acc1 = {}, acc2 = {};
#pragma unroll 2
    for (int kk = 0; kk < DD; kk += 32) {
        f16x8 b0 = *(const f16x8*)(Bp + kk);
        f16x8 b1 = *(const f16x8*)(Bp + kk + 16);
        v16h  bb = SHUF16(b0, b1);

        f16x8 a00 = *(const f16x8*)(Ar0 + kk);
        f16x8 a01 = *(const f16x8*)(Ar0 + kk + 16);
        v16h  aa0 = SHUF16(a00, a01);
        acc0 = __builtin_amdgcn_wmma_f32_16x16x32_f16(false, aa0, false, bb,
                                                      (short)0, acc0, false, false);

        f16x8 a10 = *(const f16x8*)(Ar1 + kk);
        f16x8 a11 = *(const f16x8*)(Ar1 + kk + 16);
        v16h  aa1 = SHUF16(a10, a11);
        acc1 = __builtin_amdgcn_wmma_f32_16x16x32_f16(false, aa1, false, bb,
                                                      (short)0, acc1, false, false);

        f16x8 a20 = *(const f16x8*)(Ar2 + kk);
        f16x8 a21 = *(const f16x8*)(Ar2 + kk + 16);
        v16h  aa2 = SHUF16(a20, a21);
        acc2 = __builtin_amdgcn_wmma_f32_16x16x32_f16(false, aa2, false, bb,
                                                      (short)0, acc2, false, false);
    }

    // C/D layout: VGPR j -> M = j (lanes 0-15) / j+8 (lanes 16-31); N = lane%16
    {
        const int colS = 16 * wv + m;
        const int rb   = 8 * hf;
#pragma unroll
        for (int j = 0; j < 8; ++j) {
            S[( 0 + rb + j) * LP + colS] = acc0[j];
            S[(16 + rb + j) * LP + colS] = acc1[j];
            S[(32 + rb + j) * LP + colS] = acc2[j];
        }
    }
    __syncthreads();

    // ---- Sinkhorn on the 48x64 tile in LDS (matches reference order) ----
    const int nR = img_lens[i];
    const int nW = cap_lens[c];

    // P = mask ? exp((S-1)/lambda) : 0 ; accumulate total
    float part = 0.f;
    for (int idx = tid; idx < RP * WP; idx += 128) {
        int r = idx >> 6, w = idx & 63;
        float s = S[r * LP + w];
        float p = (r < nR && w < nW) ? __expf((s - 1.0f) * LAMB_INV) : 0.0f;
        P[r * LP + w] = p;
        part += p;
    }
    red[tid] = part;
    __syncthreads();
    for (int s = 64; s > 0; s >>= 1) {
        if (tid < s) red[tid] += red[tid + s];
        __syncthreads();
    }
    float scale = 1.0f / (red[0] + EPSV);
    __syncthreads();
    for (int idx = tid; idx < RP * WP; idx += 128) {
        int r = idx >> 6, w = idx & 63;
        P[r * LP + w] *= scale;
    }
    __syncthreads();

    const float rmarg = 1.0f / (float)nR;
    const float cmarg = 1.0f / (float)nW;
    for (int it = 0; it < 3; ++it) {
        if (tid < RP) {
            float u = 0.f;
            for (int w = 0; w < WP; ++w) u += P[tid * LP + w];
            rs[tid] = ((tid < nR) ? rmarg : 0.0f) / (u + EPSV);
        }
        __syncthreads();
        for (int idx = tid; idx < RP * WP; idx += 128) {
            int r = idx >> 6, w = idx & 63;
            P[r * LP + w] *= rs[r];
        }
        __syncthreads();
        if (tid < WP) {
            float v = 0.f;
            for (int r = 0; r < RP; ++r) v += P[r * LP + tid];
            cs[tid] = ((tid < nW) ? cmarg : 0.0f) / (v + EPSV);
        }
        __syncthreads();
        for (int idx = tid; idx < RP * WP; idx += 128) {
            int r = idx >> 6, w = idx & 63;
            P[r * LP + w] *= cs[w];
        }
        __syncthreads();
    }

    // sims = sum(S * P) (P already zero outside mask)
    float accv = 0.f;
    for (int idx = tid; idx < RP * WP; idx += 128) {
        int r = idx >> 6, w = idx & 63;
        accv += S[r * LP + w] * P[r * LP + w];
    }
    red[tid] = accv;
    __syncthreads();
    for (int s = 64; s > 0; s >>= 1) {
        if (tid < s) red[tid] += red[tid + s];
        __syncthreads();
    }
    if (tid == 0) out[(size_t)i * BT + c] = red[0];
}

extern "C" void kernel_launch(void* const* d_in, const int* in_sizes, int n_in,
                              void* d_out, int out_size, void* d_ws, size_t ws_size,
                              hipStream_t stream) {
    const float* imgs = (const float*)d_in[0];   // [128,36,1024] f32
    const float* caps = (const float*)d_in[1];   // [128,50,1024] f32
    const int*   il   = (const int*)d_in[2];     // [128]
    const int*   cl   = (const int*)d_in[3];     // [128]
    float*       out  = (float*)d_out;           // [128,128] f32

    _Float16* A16 = (_Float16*)d_ws;                                   // 128*48*1024 halves
    _Float16* B16 = (_Float16*)((char*)d_ws + (size_t)BI * RP * DD * 2);

    {
        long t4 = (long)BI * RP * (DD / 4);
        cvt_pad_f16<<<(int)((t4 + 255) / 256), 256, 0, stream>>>(imgs, A16, RR, RP);
    }
    {
        long t4 = (long)BT * WP * (DD / 4);
        cvt_pad_f16<<<(int)((t4 + 255) / 256), 256, 0, stream>>>(caps, B16, WW, WP);
    }

    dim3 grid(BT, BI);
    sinkhorn_wmma<<<grid, 128, 0, stream>>>(A16, B16, il, cl, out);
}
